// HeterogeneousRelationFusion_37580963840549
// MI455X (gfx1250) — matrix operand order repaired and checked by reading, not compiled, not run on hardware
//
#include <hip/hip_runtime.h>
#include <hip/hip_bf16.h>
#include <math.h>

// ---------------------------------------------------------------------------
// RGAT fused forward for MI455X (gfx1250, wave32, WMMA).
//
// Phases:
//  0) pack: x -> f16, weights -> WMMA-B swizzled f16 fragments
//  1) QKV projection per relation:  (N x 128) @ (128 x 128), f16 WMMA, f32 acc
//  2) edge logits + segment max  (uint-encoded float atomicMax)
//  3) exp + segment sum          (f32 atomicAdd)
//  4) attn * V scatter           (f32 atomicAdd)
//  5) out-projection WMMA + bias + residual + LayerNorm
// ---------------------------------------------------------------------------

#define N_NODES   50000
#define N_EDGES   640000
#define HIDDEN    128
#define N_REL     5
#define N_HEADS   8
#define HEAD_DIM  16
#define KTILES    4          // 128 / 32
#define MTILES    (N_NODES / 16)   // 3125, exact

typedef __attribute__((ext_vector_type(16))) _Float16 v16h;
typedef __attribute__((ext_vector_type(8)))  _Float16 v8h;
typedef __attribute__((ext_vector_type(8)))  float    v8f;

// order-preserving float <-> uint map for atomicMax on signed floats
__device__ __forceinline__ unsigned f2ord(float f) {
    unsigned u = __float_as_uint(f);
    return (u >> 31) ? ~u : (u | 0x80000000u);
}
__device__ __forceinline__ float ord2f(unsigned u) {
    return (u & 0x80000000u) ? __uint_as_float(u ^ 0x80000000u)
                             : __uint_as_float(~u);
}

__device__ __forceinline__ v8f wmma_f16(v16h a, v16h b, v8f c) {
    return __builtin_amdgcn_wmma_f32_16x16x32_f16(
        /*neg_a=*/false, a, /*neg_b=*/false, b,
        /*c_mod=*/(short)0, c, /*reuse_a=*/false, /*reuse_b=*/false);
}

// ---------------------------------------------------------------------------
// Kernel: f32 -> f16 copy of node embeddings
__global__ void k_pack_x(const float* __restrict__ x, _Float16* __restrict__ xh, int n) {
    int gid = blockIdx.x * blockDim.x + threadIdx.x;
    if (gid < n) xh[gid] = (_Float16)x[gid];
}

// ---------------------------------------------------------------------------
// Kernel: pack W_Q/W_K/W_V (R,8,128,16) and out_proj_w (128,128, used transposed)
// into WMMA B fragments: frag layout = [frag][lane 0..31][t 0..15] halves, where
// lane holds column n = lane%16 and K rows (t<8 ? 8*(lane/16)+t : 16+8*(lane/16)+t-8) of a 32x16 tile.
// frags 0..479  : ((r*3+m)*8 + ct)*4 + kt   for QKV
// frags 480..511: 480 + ct*4 + kt           for out_proj (B[k][n] = W[n][k])
__global__ void k_pack_w(const float* __restrict__ WQ, const float* __restrict__ WK,
                         const float* __restrict__ WV, const float* __restrict__ WO,
                         _Float16* __restrict__ wpk) {
    int gid = blockIdx.x * blockDim.x + threadIdx.x;   // 512*512 threads
    int f    = gid >> 9;
    int lane = (gid >> 4) & 31;
    int t    = gid & 15;
    int n    = lane & 15;
    int koff = (t < 8) ? ((lane >> 4) * 8 + t) : (16 + (lane >> 4) * 8 + (t - 8));
    float val;
    if (f < 480) {
        int kt = f & 3;
        int f4 = f >> 2;
        int ct = f4 & 7;          // head
        int rm = f4 >> 3;
        int m  = rm % 3;          // 0=Q 1=K 2=V
        int r  = rm / 3;
        const float* W = (m == 0) ? WQ : (m == 1) ? WK : WV;
        // W[r][ct][hid][d], hid = kt*32+koff, d = n
        val = W[(((size_t)(r * N_HEADS + ct)) * HIDDEN + (kt * 32 + koff)) * HEAD_DIM + n];
    } else {
        int fo = f - 480;
        int kt = fo & 3;
        int ct = fo >> 2;
        // B[k][n] = out_proj_w[n_global][k_global]
        val = WO[(size_t)(ct * 16 + n) * HIDDEN + (kt * 32 + koff)];
    }
    wpk[(size_t)f * 512 + lane * 16 + t] = (_Float16)val;
}

// ---------------------------------------------------------------------------
// Kernel: init accumulators (attn_out=0, sum=0, max=ord(-1e9))
__global__ void k_init_attn(float* __restrict__ attn, int n) {
    int gid = blockIdx.x * blockDim.x + threadIdx.x;
    if (gid < n) attn[gid] = 0.0f;
}
__global__ void k_init_rh(unsigned* __restrict__ mx, float* __restrict__ sm, int n) {
    int gid = blockIdx.x * blockDim.x + threadIdx.x;
    if (gid < n) { mx[gid] = f2ord(-1e9f); sm[gid] = 0.0f; }
}

// ---------------------------------------------------------------------------
// Kernel 1: per-relation Q/K/V projection.
// Block = 256 threads = 8 waves; block owns a 16-row tile, wave w owns 16-col
// tile w. Each wave runs 5 rel * 3 mat * 4 ktiles = 60 WMMAs.
__global__ void __launch_bounds__(256)
k_proj(const _Float16* __restrict__ xh, const _Float16* __restrict__ wpk,
       _Float16* __restrict__ Qp, _Float16* __restrict__ Kp, _Float16* __restrict__ Vp) {
    const int mtile = blockIdx.x;
    const int wave  = threadIdx.x >> 5;
    const int lane  = threadIdx.x & 31;
    const int mrow  = mtile * 16 + (lane & 15);
    const int s1    = (lane >> 4) * 8;     // A K-chunk start: 0 or 8

    // A fragments: 16x32 f16; lane(<16) holds row, K in [s1,s1+8) and [s1+16,s1+24)
    v16h a[KTILES];
    const _Float16* xrow = xh + (size_t)mrow * HIDDEN;
#pragma unroll
    for (int kt = 0; kt < KTILES; ++kt) {
        v8h lo = *(const v8h*)(xrow + kt * 32 + s1);
        v8h hi = *(const v8h*)(xrow + kt * 32 + s1 + 16);
        a[kt] = __builtin_shufflevector(lo, hi, 0, 1, 2, 3, 4, 5, 6, 7,
                                                8, 9, 10, 11, 12, 13, 14, 15);
    }

    _Float16* outs[3] = {Qp, Kp, Vp};
    const int rtile = (lane >> 4) * 8;     // C row offset for this lane half

    for (int r = 0; r < N_REL; ++r) {
#pragma unroll
        for (int m = 0; m < 3; ++m) {
            v8f c = {};
            const _Float16* bb =
                wpk + (size_t)(((r * 3 + m) * 8 + wave) * 4) * 512;
#pragma unroll
            for (int kt = 0; kt < KTILES; ++kt) {
                v16h b = *(const v16h*)(bb + (size_t)kt * 512 + lane * 16);
                c = wmma_f16(a[kt], b, c);
            }
            _Float16* o = outs[m] +
                ((size_t)r * N_NODES + mtile * 16) * HIDDEN + wave * 16;
#pragma unroll
            for (int i = 0; i < 8; ++i)
                o[(size_t)(i + rtile) * HIDDEN + (lane & 15)] = (_Float16)c[i];
        }
    }
}

// ---------------------------------------------------------------------------
// Kernel 2: edge logits + segment max. One thread per (edge, head).
__global__ void k_logits(const _Float16* __restrict__ Qp, const _Float16* __restrict__ Kp,
                         const int* __restrict__ ei, const int* __restrict__ et,
                         const float* __restrict__ bias,
                         float* __restrict__ logits, unsigned* __restrict__ maxbuf) {
    int gid = blockIdx.x * blockDim.x + threadIdx.x;
    int e = gid >> 3, h = gid & 7;
    if (e >= N_EDGES) return;
    int src = ei[e], dst = ei[N_EDGES + e], r = et[e];
    const v16h q = *(const v16h*)(Qp + ((size_t)r * N_NODES + dst) * HIDDEN + h * HEAD_DIM);
    const v16h k = *(const v16h*)(Kp + ((size_t)r * N_NODES + src) * HIDDEN + h * HEAD_DIM);
    float acc = 0.0f;
#pragma unroll
    for (int i = 0; i < HEAD_DIM; ++i) acc += (float)q[i] * (float)k[i];
    float lg = acc * 0.25f + bias[r * N_HEADS + h];   // 1/sqrt(16)
    logits[(size_t)e * N_HEADS + h] = lg;
    atomicMax(&maxbuf[((size_t)r * N_NODES + dst) * N_HEADS + h], f2ord(lg));
}

// ---------------------------------------------------------------------------
// Kernel 3: exp(logit - max) + segment sum.
__global__ void k_expsum(const int* __restrict__ ei, const int* __restrict__ et,
                         float* __restrict__ logits, const unsigned* __restrict__ maxbuf,
                         float* __restrict__ sumbuf) {
    int gid = blockIdx.x * blockDim.x + threadIdx.x;
    int e = gid >> 3, h = gid & 7;
    if (e >= N_EDGES) return;
    int dst = ei[N_EDGES + e], r = et[e];
    size_t rh = ((size_t)r * N_NODES + dst) * N_HEADS + h;
    float m  = ord2f(maxbuf[rh]);
    float el = __expf(logits[(size_t)e * N_HEADS + h] - m);
    logits[(size_t)e * N_HEADS + h] = el;
    atomicAdd(&sumbuf[rh], el);
}

// ---------------------------------------------------------------------------
// Kernel 4: attn-weighted V scatter into f32 accumulator.
__global__ void k_accum(const _Float16* __restrict__ Vp,
                        const int* __restrict__ ei, const int* __restrict__ et,
                        const float* __restrict__ logits, const float* __restrict__ sumbuf,
                        float* __restrict__ attn) {
    int gid = blockIdx.x * blockDim.x + threadIdx.x;
    int e = gid >> 3, h = gid & 7;
    if (e >= N_EDGES) return;
    int src = ei[e], dst = ei[N_EDGES + e], r = et[e];
    size_t rh = ((size_t)r * N_NODES + dst) * N_HEADS + h;
    float w = logits[(size_t)e * N_HEADS + h] / (sumbuf[rh] + 1e-12f);
    const v16h v = *(const v16h*)(Vp + ((size_t)r * N_NODES + src) * HIDDEN + h * HEAD_DIM);
    float* o = attn + (size_t)dst * HIDDEN + h * HEAD_DIM;
#pragma unroll
    for (int d = 0; d < HEAD_DIM; ++d) atomicAdd(&o[d], w * (float)v[d]);
}

// ---------------------------------------------------------------------------
// Kernel 5: out = LN(attn @ Wo^T + b + x). WMMA out-projection, then LN in LDS.
__global__ void __launch_bounds__(256)
k_final(const float* __restrict__ attn, const _Float16* __restrict__ wpk,
        const float* __restrict__ x, const float* __restrict__ ob,
        const float* __restrict__ gamma, const float* __restrict__ beta,
        float* __restrict__ out) {
    __shared__ float sh[16][HIDDEN];
    __shared__ float smu[16], srs[16];

    const int mtile = blockIdx.x;
    const int wave  = threadIdx.x >> 5;
    const int lane  = threadIdx.x & 31;
    const int mrow  = mtile * 16 + (lane & 15);
    const int s1    = (lane >> 4) * 8;

    // A fragments from f32 accumulator (convert in-register)
    v16h a[KTILES];
    const float* arow = attn + (size_t)mrow * HIDDEN;
#pragma unroll
    for (int kt = 0; kt < KTILES; ++kt) {
        v16h t;
#pragma unroll
        for (int i = 0; i < 8; ++i) {
            t[i]     = (_Float16)arow[kt * 32 + s1 + i];
            t[i + 8] = (_Float16)arow[kt * 32 + s1 + 16 + i];
        }
        a[kt] = t;
    }

    v8f c = {};
    const _Float16* bb = wpk + (size_t)(480 + wave * 4) * 512;
#pragma unroll
    for (int kt = 0; kt < KTILES; ++kt) {
        v16h b = *(const v16h*)(bb + (size_t)kt * 512 + lane * 16);
        c = wmma_f16(a[kt], b, c);
    }

    // h = proj + bias + residual -> LDS
    const int rtile = (lane >> 4) * 8;
    const int col   = wave * 16 + (lane & 15);
#pragma unroll
    for (int i = 0; i < 8; ++i) {
        int rr = i + rtile;
        sh[rr][col] = c[i] + ob[col] + x[(size_t)(mtile * 16 + rr) * HIDDEN + col];
    }
    __syncthreads();

    if (threadIdx.x < 16) {
        float s = 0.0f, s2 = 0.0f;
#pragma unroll 8
        for (int j = 0; j < HIDDEN; ++j) {
            float v = sh[threadIdx.x][j];
            s += v; s2 += v * v;
        }
        float mu  = s * (1.0f / HIDDEN);
        float var = s2 * (1.0f / HIDDEN) - mu * mu;
        smu[threadIdx.x] = mu;
        srs[threadIdx.x] = rsqrtf(var + 1e-5f);
    }
    __syncthreads();

    int row = threadIdx.x >> 4;
    int c0  = (threadIdx.x & 15) * 8;
    float mu = smu[row], rs = srs[row];
#pragma unroll
    for (int j = 0; j < 8; ++j) {
        int cc = c0 + j;
        out[(size_t)(mtile * 16 + row) * HIDDEN + cc] =
            (sh[row][cc] - mu) * rs * gamma[cc] + beta[cc];
    }
}

// ---------------------------------------------------------------------------
extern "C" void kernel_launch(void* const* d_in, const int* in_sizes, int n_in,
                              void* d_out, int out_size, void* d_ws, size_t ws_size,
                              hipStream_t stream) {
    const float* x    = (const float*)d_in[0];
    const int*   ei   = (const int*)d_in[1];   // (2, E)
    const int*   et   = (const int*)d_in[2];   // (E,)
    const float* WQ   = (const float*)d_in[3];
    const float* WK   = (const float*)d_in[4];
    const float* WV   = (const float*)d_in[5];
    const float* bias = (const float*)d_in[6];
    const float* WO   = (const float*)d_in[7];
    const float* ob   = (const float*)d_in[8];
    const float* gam  = (const float*)d_in[9];
    const float* bet  = (const float*)d_in[10];
    float* out = (float*)d_out;

    // workspace carve (all sizes multiple of 256B)
    char* ws = (char*)d_ws;
    size_t off = 0;
    auto carve = [&](size_t bytes) {
        void* p = ws + off;
        off += (bytes + 255) & ~(size_t)255;
        return p;
    };
    _Float16* Xh  = (_Float16*)carve((size_t)N_NODES * HIDDEN * 2);
    _Float16* Wpk = (_Float16*)carve((size_t)512 * 512 * 2);
    _Float16* Qp  = (_Float16*)carve((size_t)N_REL * N_NODES * HIDDEN * 2);
    _Float16* Kp  = (_Float16*)carve((size_t)N_REL * N_NODES * HIDDEN * 2);
    _Float16* Vp  = (_Float16*)carve((size_t)N_REL * N_NODES * HIDDEN * 2);
    float*    lg  = (float*)carve((size_t)N_EDGES * N_HEADS * 4);
    unsigned* mxb = (unsigned*)carve((size_t)N_REL * N_NODES * N_HEADS * 4);
    float*    smb = (float*)carve((size_t)N_REL * N_NODES * N_HEADS * 4);
    float*    att = (float*)carve((size_t)N_NODES * HIDDEN * 4);

    const int B = 256;
    // phase 0: packing + init
    k_pack_x<<<(N_NODES * HIDDEN + B - 1) / B, B, 0, stream>>>(x, Xh, N_NODES * HIDDEN);
    k_pack_w<<<(512 * 512) / B, B, 0, stream>>>(WQ, WK, WV, WO, Wpk);
    k_init_attn<<<(N_NODES * HIDDEN + B - 1) / B, B, 0, stream>>>(att, N_NODES * HIDDEN);
    k_init_rh<<<(N_REL * N_NODES * N_HEADS + B - 1) / B, B, 0, stream>>>(
        mxb, smb, N_REL * N_NODES * N_HEADS);

    // phase 1: QKV projections (WMMA)
    k_proj<<<MTILES, B, 0, stream>>>(Xh, Wpk, Qp, Kp, Vp);

    // phases 2-4: edge softmax + aggregation
    const int EG = (N_EDGES * N_HEADS) / B;  // exact: 20000
    k_logits<<<EG, B, 0, stream>>>(Qp, Kp, ei, et, bias, lg, mxb);
    k_expsum<<<EG, B, 0, stream>>>(ei, et, lg, mxb, smb);
    k_accum <<<EG, B, 0, stream>>>(Vp, ei, et, lg, smb, att);

    // phase 5: out-projection (WMMA) + residual + LayerNorm
    k_final<<<MTILES, B, 0, stream>>>(att, Wpk, x, ob, gam, bet, out);

    (void)in_sizes; (void)n_in; (void)out_size; (void)ws_size;
}
